// ExpertGroup_75488345195331
// MI455X (gfx1250) — matrix operand order repaired and checked by reading, not compiled, not run on hardware
//
#include <hip/hip_runtime.h>
#include <hip/hip_bf16.h>
#include <math.h>

#define IN_DIM  2048
#define HID_DIM 4096
#define OUT_DIM 2048
#define M_BLOCK 32
#define HP      (HID_DIM + 8)   // LDS pitch in bf16 elements

typedef __attribute__((ext_vector_type(16))) __bf16 v16bf;
typedef __attribute__((ext_vector_type(8)))  __bf16 v8bf;
typedef __attribute__((ext_vector_type(8)))  float  v8f;
typedef __attribute__((ext_vector_type(4)))  float  f32x4;

union FragBF {
    v16bf v;
    v8bf  h[2];
    __bf16 e[16];
};

// 16-bf16 fragment from two 8-float chunks (f32 source, convert on the fly).
__device__ __forceinline__ v16bf frag_from_f32(const float* __restrict__ p0,
                                               const float* __restrict__ p1) {
    f32x4 a0 = *(const f32x4*)(p0);
    f32x4 a1 = *(const f32x4*)(p0 + 4);
    f32x4 b0 = *(const f32x4*)(p1);
    f32x4 b1 = *(const f32x4*)(p1 + 4);
    FragBF f;
#pragma unroll
    for (int i = 0; i < 4; ++i) {
        f.e[i]      = (__bf16)a0[i];
        f.e[4 + i]  = (__bf16)a1[i];
        f.e[8 + i]  = (__bf16)b0[i];
        f.e[12 + i] = (__bf16)b1[i];
    }
    return f.v;
}

// A-fragment (bf16 source): 16 contiguous bf16 per lane (chunks at p and p+8).
__device__ __forceinline__ v16bf afrag_bf16(const __bf16* __restrict__ p) {
    FragBF f;
    f.h[0] = *(const v8bf*)(p);
    f.h[1] = *(const v8bf*)(p + 8);
    return f.v;
}

// B-fragment (bf16 source): 16 contiguous bf16 per lane (same k order as afrag_bf16).
__device__ __forceinline__ v16bf bfrag_bf16(const __bf16* __restrict__ p) {
    FragBF f;
    f.h[0] = *(const v8bf*)(p);
    f.h[1] = *(const v8bf*)(p + 8);
    return f.v;
}

__device__ __forceinline__ v8f wmma_bf16(v16bf a, v16bf b, v8f c) {
    return __builtin_amdgcn_wmma_f32_16x16x32_bf16(false, a, false, b, (short)0, c,
                                                   false, false);
}

__device__ __forceinline__ float gelu_exact(float v) {
    return 0.5f * v * (1.0f + erff(v * 0.70710678118654752f));
}

// ---------------------------------------------------------------------------
// Pass 0 (PRE path): convert x, W1[e], W2[e] from f32 to bf16 into workspace.
// Pure streaming: ~235 MB of HBM traffic, ~10 us at 23.3 TB/s.
// ---------------------------------------------------------------------------
__global__ __launch_bounds__(256)
void convert_to_bf16(const int* __restrict__ expert_idx,
                     const float* __restrict__ x,
                     const float* __restrict__ W1,
                     const float* __restrict__ W2,
                     __bf16* __restrict__ xb,
                     __bf16* __restrict__ w1b,
                     __bf16* __restrict__ w2b,
                     long n_x) {
    const long S1 = (long)HID_DIM * IN_DIM;
    const long S2 = (long)OUT_DIM * HID_DIM;
    const int  e  = *expert_idx;
    const float* W1e = W1 + (size_t)e * S1;
    const float* W2e = W2 + (size_t)e * S2;

    const long total8 = (n_x + S1 + S2) >> 3;   // all sizes divisible by 8
    for (long i = (long)blockIdx.x * blockDim.x + threadIdx.x; i < total8;
         i += (long)gridDim.x * blockDim.x) {
        const long off = i << 3;
        const float* src;
        __bf16*      dst;
        if (off < n_x)           { src = x   + off;            dst = xb  + off; }
        else if (off < n_x + S1) { src = W1e + (off - n_x);    dst = w1b + (off - n_x); }
        else                     { src = W2e + (off - n_x - S1); dst = w2b + (off - n_x - S1); }
        f32x4 a = *(const f32x4*)(src);
        f32x4 b = *(const f32x4*)(src + 4);
        v8bf o;
#pragma unroll
        for (int j = 0; j < 4; ++j) {
            o[j]     = (__bf16)a[j];
            o[4 + j] = (__bf16)b[j];
        }
        *(v8bf*)dst = o;
    }
}

// ---------------------------------------------------------------------------
// Fused expert FFN. PRE=true: operands pre-converted to bf16 in workspace
// (hot loop = pure bf16 loads + WMMA). PRE=false: convert f32 on the fly.
// Each block owns 32 rows; h = gelu(x @ W1^T) lives in LDS (bf16, 32x4096).
// Wave macro-tile: 32(M) x 64(N) = 2x4 WMMA tiles, 8 accumulators.
// ---------------------------------------------------------------------------
template <bool PRE>
__global__ __launch_bounds__(256)
void expert_ffn_fused(const int* __restrict__ expert_idx,
                      const float* __restrict__ x,
                      const float* __restrict__ W1,
                      const float* __restrict__ W2,
                      const __bf16* __restrict__ xb,
                      const __bf16* __restrict__ w1b,
                      const __bf16* __restrict__ w2b,
                      float* __restrict__ out) {
    extern __shared__ __bf16 h_lds[];   // [M_BLOCK][HP]

    const int lane = threadIdx.x & 31;
    const int wave = threadIdx.x >> 5;
    const int lm   = lane & 15;
    const int hi   = lane >> 4;
    const int row0 = blockIdx.x * M_BLOCK;

    const float* W1e = nullptr;
    const float* W2e = nullptr;
    if constexpr (!PRE) {
        const int e = *expert_idx;
        W1e = W1 + (size_t)e * (size_t)HID_DIM * IN_DIM;
        W2e = W2 + (size_t)e * (size_t)OUT_DIM * HID_DIM;
    }

    // ---------------- Phase 1: h = gelu(x @ W1^T) -> LDS (bf16) ----------------
    const float*  xr0f = nullptr; const float*  xr1f = nullptr;
    const __bf16* xr0b = nullptr; const __bf16* xr1b = nullptr;
    if constexpr (PRE) {
        xr0b = xb + (size_t)(row0 + lm) * IN_DIM;
        xr1b = xb + (size_t)(row0 + 16 + lm) * IN_DIM;
    } else {
        xr0f = x + (size_t)(row0 + lm) * IN_DIM;
        xr1f = x + (size_t)(row0 + 16 + lm) * IN_DIM;
    }

    for (int g = 0; g < (HID_DIM / 16) / 32; ++g) {            // 8 groups
        const int nt = (g * 8 + wave) * 4;                     // 4 adjacent N-tiles
        const __bf16* wrb[4]; const float* wrf[4];
#pragma unroll
        for (int j = 0; j < 4; ++j) {
            if constexpr (PRE) wrb[j] = w1b + (size_t)((nt + j) * 16 + lm) * IN_DIM;
            else               wrf[j] = W1e + (size_t)((nt + j) * 16 + lm) * IN_DIM;
        }
        v8f acc[2][4] = {};
#pragma unroll 2
        for (int k0 = 0; k0 < IN_DIM; k0 += 32) {
            const int ka = k0 + hi * 16;
            const int kb = k0 + hi * 16;
            v16bf a0, a1, bfr[4];
            if constexpr (PRE) {
                a0 = afrag_bf16(xr0b + ka);
                a1 = afrag_bf16(xr1b + ka);
#pragma unroll
                for (int j = 0; j < 4; ++j) {
                    __builtin_prefetch(wrb[j] + kb + 256, 0, 1);
                    bfr[j] = bfrag_bf16(wrb[j] + kb);
                }
            } else {
                a0 = frag_from_f32(xr0f + ka, xr0f + ka + 8);
                a1 = frag_from_f32(xr1f + ka, xr1f + ka + 8);
#pragma unroll
                for (int j = 0; j < 4; ++j) {
                    __builtin_prefetch(wrf[j] + kb + 128, 0, 1);
                    bfr[j] = frag_from_f32(wrf[j] + kb, wrf[j] + kb + 8);
                }
            }
#pragma unroll
            for (int j = 0; j < 4; ++j) {
                acc[0][j] = wmma_bf16(a0, bfr[j], acc[0][j]);
                acc[1][j] = wmma_bf16(a1, bfr[j], acc[1][j]);
            }
        }
        // C layout: VGPR r -> row (r + hi*8), col lm. Exact-erf GELU, store bf16.
#pragma unroll
        for (int j = 0; j < 4; ++j) {
#pragma unroll
            for (int r = 0; r < 8; ++r) {
                const int m0 = r + hi * 8;
                const int m1 = 16 + r + hi * 8;
                h_lds[(size_t)m0 * HP + (nt + j) * 16 + lm] = (__bf16)gelu_exact(acc[0][j][r]);
                h_lds[(size_t)m1 * HP + (nt + j) * 16 + lm] = (__bf16)gelu_exact(acc[1][j][r]);
            }
        }
    }

    __syncthreads();

    // ---------------- Phase 2: out = h @ W2^T (A from LDS) ----------------
    const __bf16* hr0 = h_lds + (size_t)lm * HP;
    const __bf16* hr1 = h_lds + (size_t)(16 + lm) * HP;

    for (int g = 0; g < (OUT_DIM / 16) / 32; ++g) {            // 4 groups
        const int nt = (g * 8 + wave) * 4;
        const __bf16* wrb[4]; const float* wrf[4];
#pragma unroll
        for (int j = 0; j < 4; ++j) {
            if constexpr (PRE) wrb[j] = w2b + (size_t)((nt + j) * 16 + lm) * HID_DIM;
            else               wrf[j] = W2e + (size_t)((nt + j) * 16 + lm) * HID_DIM;
        }
        v8f acc[2][4] = {};
#pragma unroll 2
        for (int k0 = 0; k0 < HID_DIM; k0 += 32) {
            const int ka = k0 + hi * 16;
            const int kb = k0 + hi * 16;
            v16bf a0 = afrag_bf16(hr0 + ka);
            v16bf a1 = afrag_bf16(hr1 + ka);
            v16bf bfr[4];
            if constexpr (PRE) {
#pragma unroll
                for (int j = 0; j < 4; ++j) {
                    __builtin_prefetch(wrb[j] + kb + 256, 0, 1);
                    bfr[j] = bfrag_bf16(wrb[j] + kb);
                }
            } else {
#pragma unroll
                for (int j = 0; j < 4; ++j) {
                    __builtin_prefetch(wrf[j] + kb + 128, 0, 1);
                    bfr[j] = frag_from_f32(wrf[j] + kb, wrf[j] + kb + 8);
                }
            }
#pragma unroll
            for (int j = 0; j < 4; ++j) {
                acc[0][j] = wmma_bf16(a0, bfr[j], acc[0][j]);
                acc[1][j] = wmma_bf16(a1, bfr[j], acc[1][j]);
            }
        }
        float* o = out + (size_t)row0 * OUT_DIM;
#pragma unroll
        for (int j = 0; j < 4; ++j) {
#pragma unroll
            for (int r = 0; r < 8; ++r) {
                const int m0 = r + hi * 8;
                const int m1 = 16 + r + hi * 8;
                o[(size_t)m0 * OUT_DIM + (nt + j) * 16 + lm] = acc[0][j][r];
                o[(size_t)m1 * OUT_DIM + (nt + j) * 16 + lm] = acc[1][j][r];
            }
        }
    }
}

extern "C" void kernel_launch(void* const* d_in, const int* in_sizes, int n_in,
                              void* d_out, int out_size, void* d_ws, size_t ws_size,
                              hipStream_t stream) {
    (void)n_in; (void)out_size;
    const int*   eidx = (const int*)d_in[0];
    const float* x    = (const float*)d_in[1];
    const float* W1   = (const float*)d_in[2];
    const float* W2   = (const float*)d_in[3];
    float*       out  = (float*)d_out;

    const long   n_x     = (long)in_sizes[1];            // 16384*2048
    const int    n_nodes = (int)(n_x / IN_DIM);
    const int    nblocks = n_nodes / M_BLOCK;            // 512
    const size_t smem    = (size_t)M_BLOCK * HP * sizeof(__bf16);

    const size_t S1 = (size_t)HID_DIM * IN_DIM;
    const size_t S2 = (size_t)OUT_DIM * HID_DIM;
    const size_t need = ((size_t)n_x + S1 + S2) * sizeof(__bf16);   // ~100.7 MB

    if (ws_size >= need) {
        __bf16* xb  = (__bf16*)d_ws;
        __bf16* w1b = xb + n_x;
        __bf16* w2b = w1b + S1;
        convert_to_bf16<<<4096, 256, 0, stream>>>(eidx, x, W1, W2, xb, w1b, w2b, n_x);
        expert_ffn_fused<true><<<nblocks, 256, smem, stream>>>(
            eidx, x, W1, W2, xb, w1b, w2b, out);
    } else {
        expert_ffn_fused<false><<<nblocks, 256, smem, stream>>>(
            eidx, x, W1, W2, nullptr, nullptr, nullptr, out);
    }
}
